// TestModule_2345052143639
// MI455X (gfx1250) — matrix operand order repaired and checked by reading, not compiled, not run on hardware
//
#include <hip/hip_runtime.h>
#include <hip/hip_bf16.h>

typedef __attribute__((ext_vector_type(16))) __bf16       v16bf;
typedef __attribute__((ext_vector_type(8)))  float        v8f;
typedef __attribute__((ext_vector_type(4)))  unsigned int v4u;
typedef __attribute__((ext_vector_type(8)))  int          v8i_;
typedef __attribute__((ext_vector_type(4)))  int          v4i_;

#define ATTN_B 2
#define ATTN_S 2048
#define ATTN_H 32
#define ATTN_D 128

#define KTILE   32          // keys staged per LDS tile
#define KPITCH  132         // fp32 staging pitch: 128 + 4 dword pad (TDM pad)
#define BPITCH  136         // bf16 compute pitch: 128 + 8 elem pad (bank skew)

#if __has_builtin(__builtin_amdgcn_tensor_load_to_lds)
#define USE_TDM 1
#else
#define USE_TDM 0
#endif

#if __has_builtin(__builtin_amdgcn_exp2f)
#define EXP2(x) __builtin_amdgcn_exp2f(x)
#else
#define EXP2(x) exp2f(x)
#endif

// ---------------------------------------------------------------------------
__global__ void zero_out_kernel(float* out) {
    if (threadIdx.x == 0 && blockIdx.x == 0) out[0] = 0.0f;
}

// ---------------------------------------------------------------------------
// w[b,h,k] = sum_d v[b,k,h,d].  One row per wave; lane reads float4.
// ---------------------------------------------------------------------------
__global__ __launch_bounds__(256) void wsum_kernel(const float* __restrict__ v,
                                                   float* __restrict__ w) {
    constexpr int S = ATTN_S, H = ATTN_H, D = ATTN_D;
    const int wave = threadIdx.x >> 5;
    const int lane = threadIdx.x & 31;
    const long long row = (long long)blockIdx.x * 8 + wave;   // flat (b*S+k)*H+h
    const float4* src = (const float4*)(v + row * D);
    float4 x = src[lane];
    float s = x.x + x.y + x.z + x.w;
#pragma unroll
    for (int m = 1; m <= 16; m <<= 1) s += __shfl_xor(s, m, 32);
    if (lane == 0) {
        const int b = (int)(row / ((long long)S * H));
        const int k = (int)((row / H) % S);
        const int h = (int)(row % H);
        w[((long long)(b * H + h)) * S + k] = s;
    }
}

// ---------------------------------------------------------------------------
#if USE_TDM
// One TDM 2D tile load: KTILE rows x 128 floats, global row stride H*D floats,
// into LDS at lds_byte_off with 4-dword padding per 128 dwords (pitch 132).
__device__ inline void tdm_load_k_tile(const float* gsrc, unsigned lds_byte_off) {
    unsigned long long ga = (unsigned long long)(uintptr_t)gsrc;
    v4u g0;
    g0[0] = 1u;                                               // count=1, user D#
    g0[1] = lds_byte_off;                                     // lds_addr
    g0[2] = (unsigned)(ga & 0xffffffffu);                     // global_addr[31:0]
    g0[3] = (unsigned)((ga >> 32) & 0x1ffffffu) | (2u << 30); // addr[56:32]|type=2
    v8i_ g1;
    g1[0] = (2 << 16)      // data_size = 4B
          | (1 << 20)      // pad_enable
          | (6 << 22)      // pad_interval: every 128 dwords
          | (3 << 25);     // pad_amount: 4 dwords -> pitch 132 dwords
    g1[1] = (int)(128u << 16);             // tensor_dim0 = D = 128
    g1[2] = (int)((unsigned)ATTN_S << 16); // tensor_dim1 = S (low16)
    g1[3] = (int)(128u << 16);             // tile_dim0 = 128
    g1[4] = KTILE;                         // tile_dim1 = 32 keys, tile_dim2 = 0
    g1[5] = ATTN_H * ATTN_D;               // tensor_dim0_stride = 4096 (low32)
    g1[6] = 0; g1[7] = 0;
    v4i_ g2 = {0, 0, 0, 0};
    v4i_ g3 = {0, 0, 0, 0};
    v8i_ g4 = {0, 0, 0, 0, 0, 0, 0, 0};    // extra group unused (2D tensor)
    __builtin_amdgcn_tensor_load_to_lds(g0, g1, g2, g3, g4, 0);
}
#endif

// 32-byte LDS <-> v16bf bridging through two 16B-aligned pieces.
struct B32B { uint4 lo, hi; };
__device__ inline v16bf load_bk(const __bf16* p) {
    B32B r;
    r.lo = *(const uint4*)p;
    r.hi = *(const uint4*)(p + 8);
    return __builtin_bit_cast(v16bf, r);
}

// Unshifted base-2 softmax accumulation: log2(e)/sqrt(D) is folded into the
// Q fragments, so score fragments are already log2-domain logits and each
// element costs one bare v_exp_f32.  c[r] = row (r + 8*hi), column m16.
__device__ inline void softmax_update(const v8f& c, int diag, int m16, int hi,
                                      float wv, float* l, float* a) {
#pragma unroll
    for (int r = 0; r < 8; ++r) {
        const int qr = r + (hi ? 8 : 0);
        float s = c[r];
        s = (diag & (m16 > qr)) ? -3.0e38f : s;   // exp2 -> 0 for masked
        const float p = EXP2(s);
        l[r] += p;
        a[r] += p * wv;
    }
}

// ---------------------------------------------------------------------------
// Causal attention -> scalar sum.  Block = 256 thr (8 waves) = 256 q rows;
// wave w owns rows [qb0+32w, qb0+32w+32) as two 16-row WMMA tiles.
// Pipeline per 32-key tile: TDM fp32 stage (prefetched, TENSORcnt) ->
// one cooperative fp32->bf16 convert -> bf16 WMMA from LDS (no VALU converts).
// ---------------------------------------------------------------------------
__global__ __launch_bounds__(256) void attn_sum_kernel(const float* __restrict__ q,
                                                       const float* __restrict__ k,
                                                       const float* __restrict__ wvec,
                                                       float* __restrict__ out) {
    constexpr int S = ATTN_S, H = ATTN_H, D = ATTN_D;
    // 1/sqrt(128) * log2(e): WMMA outputs become base-2 logits directly.
    constexpr float SCALE2 = 0.08838834764831845f * 1.4426950408889634f;

#if USE_TDM
    __shared__ float  fbuf[2][KTILE * KPITCH];      // fp32 TDM staging
#endif
    __shared__ __bf16 bbuf[2][KTILE * BPITCH];      // bf16 compute tiles
    __shared__ float  red[8];

    const int h    = blockIdx.y;
    const int b    = blockIdx.z;
    const int wave = threadIdx.x >> 5;
    const int lane = threadIdx.x & 31;
    const int m16  = lane & 15;
    const int hi   = lane >> 4;

    const int qb0 = blockIdx.x * 256;
    const int qA0 = __builtin_amdgcn_readfirstlane(qb0 + wave * 32);
    const int qB0 = qA0 + 16;
    const int w0  = __builtin_amdgcn_readfirstlane(wave == 0);

    // ---- Q fragments (A 16x32 layout), SCALE2 folded into the bf16 convert.
    v16bf aq[2][4];
#pragma unroll
    for (int tq = 0; tq < 2; ++tq) {
        const float* qrow =
            q + (((long long)b * S + (qA0 + tq * 16 + m16)) * H + h) * D;
#pragma unroll
        for (int ci = 0; ci < 4; ++ci) {
            const int d0 = ci * 32 + (hi ? 16 : 0);
#pragma unroll
            for (int j = 0; j < 8; ++j)
                aq[tq][ci][j] = (__bf16)(qrow[d0 + j] * SCALE2);
#pragma unroll
            for (int j = 0; j < 8; ++j)
                aq[tq][ci][8 + j] = (__bf16)(qrow[d0 + 8 + j] * SCALE2);
        }
    }

    float lA[8], aA[8], lB[8], aB[8];
#pragma unroll
    for (int r = 0; r < 8; ++r) { lA[r] = lB[r] = 0.0f; aA[r] = aB[r] = 0.0f; }

    const float* kbase = k + (((long long)b * S) * H + h) * D;
    const float* wp    = wvec + ((long long)(b * H + h)) * S;

    const int ntiles = qb0 / KTILE + 8;     // staged 32-key tiles, causal bound

    // convert-stage mapping: 16 floats per thread (row, 16-elem segment)
    const int crow = threadIdx.x >> 3;
    const int cseg = threadIdx.x & 7;

#if USE_TDM
    if (w0) tdm_load_k_tile(kbase, (unsigned)(uintptr_t)&fbuf[0][0]);
#endif

    for (int t = 0; t < ntiles; ++t) {
        const int kb_tile = t * KTILE;
        const int buf = t & 1;

#if USE_TDM
        if (w0) __builtin_amdgcn_s_wait_tensorcnt(0);
        __syncthreads();                         // fbuf[buf] holds tile t
        if (w0 && t + 1 < ntiles)
            tdm_load_k_tile(kbase + (long long)(kb_tile + KTILE) * (H * D),
                            (unsigned)(uintptr_t)&fbuf[buf ^ 1][0]);
        const float4* csrc = (const float4*)(&fbuf[buf][crow * KPITCH + cseg * 16]);
#else
        __syncthreads();
        const float4* csrc = (const float4*)(kbase +
            (long long)(kb_tile + crow) * (H * D) + cseg * 16);
#endif
        {   // cooperative fp32 -> bf16 convert (once per block per tile)
            v16bf tb;
#pragma unroll
            for (int g = 0; g < 4; ++g) {
                float4 x = csrc[g];
                tb[4 * g + 0] = (__bf16)x.x;
                tb[4 * g + 1] = (__bf16)x.y;
                tb[4 * g + 2] = (__bf16)x.z;
                tb[4 * g + 3] = (__bf16)x.w;
            }
            B32B u = __builtin_bit_cast(B32B, tb);
            __bf16* dst = &bbuf[buf][crow * BPITCH + cseg * 16];
            *(uint4*)dst       = u.lo;
            *(uint4*)(dst + 8) = u.hi;
        }
        __syncthreads();                         // bbuf[buf] published

        const __bf16* kl = &bbuf[buf][0];

#pragma unroll
        for (int sub = 0; sub < 2; ++sub) {
            const int kb  = kb_tile + sub * 16;
            const int doB = __builtin_amdgcn_readfirstlane(kb <= qB0);
            const int doA = __builtin_amdgcn_readfirstlane(kb <= qA0);
            if (!doB) continue;                  // scalar branch, EXEC intact

            const __bf16* krow = kl + (sub * 16 + m16) * BPITCH;
            const float   wv   = wp[kb + m16];

            // Preload all 4 B-fragments so DS loads issue ahead of the
            // WMMA chain (staged dscnt waits instead of full drains).
            v16bf bk[4];
#pragma unroll
            for (int ci = 0; ci < 4; ++ci)
                bk[ci] = load_bk(krow + ci * 32 + (hi ? 16 : 0));

            if (doA) {
                v8f cA = {}, cB = {};
#pragma unroll
                for (int ci = 0; ci < 4; ++ci) {
                    cA = __builtin_amdgcn_wmma_f32_16x16x32_bf16(
                             false, aq[0][ci], false, bk[ci], (short)0, cA, false, false);
                    cB = __builtin_amdgcn_wmma_f32_16x16x32_bf16(
                             false, aq[1][ci], false, bk[ci], (short)0, cB, false, false);
                }
                softmax_update(cA, kb == qA0, m16, hi, wv, lA, aA);
                softmax_update(cB, kb == qB0, m16, hi, wv, lB, aB);
            } else {
                v8f cB = {};
#pragma unroll
                for (int ci = 0; ci < 4; ++ci) {
                    cB = __builtin_amdgcn_wmma_f32_16x16x32_bf16(
                             false, aq[1][ci], false, bk[ci], (short)0, cB, false, false);
                }
                softmax_update(cB, kb == qB0, m16, hi, wv, lB, aB);
            }
        }
    }

    // ---- Cross-lane merge (once).  Row sums live in 16-lane groups.
    float tot = 0.0f;
#pragma unroll
    for (int tq = 0; tq < 2; ++tq) {
        const float* l = tq ? lB : lA;
        const float* a = tq ? aB : aA;
#pragma unroll
        for (int r = 0; r < 8; ++r) {
            float L = l[r], A = a[r];
#pragma unroll
            for (int s = 1; s <= 8; s <<= 1) {
                L += __shfl_xor(L, s, 32);
                A += __shfl_xor(A, s, 32);
            }
            tot += A / L;
        }
    }
    tot += __shfl_xor(tot, 16, 32);   // combine row halves 0-7 / 8-15

    if (lane == 0) red[wave] = tot;
    __syncthreads();
    if (threadIdx.x == 0) {
        float s = 0.0f;
#pragma unroll
        for (int i = 0; i < 8; ++i) s += red[i];
        atomicAdd(out, s);
    }
}

// ---------------------------------------------------------------------------
extern "C" void kernel_launch(void* const* d_in, const int* in_sizes, int n_in,
                              void* d_out, int out_size, void* d_ws, size_t ws_size,
                              hipStream_t stream) {
    (void)in_sizes; (void)n_in; (void)out_size; (void)ws_size;
    const float* q = (const float*)d_in[0];
    const float* k = (const float*)d_in[1];
    const float* v = (const float*)d_in[2];
    float* out = (float*)d_out;
    float* w   = (float*)d_ws;   // B*H*S floats = 512 KB

    constexpr int B = ATTN_B, S = ATTN_S, H = ATTN_H;

    zero_out_kernel<<<1, 64, 0, stream>>>(out);

    const int rows = B * S * H;                       // 131072
    wsum_kernel<<<rows / 8, 256, 0, stream>>>(v, w);

    dim3 grid(S / 256, H, B);                         // (8, 32, 2)
    attn_sum_kernel<<<grid, 256, 0, stream>>>(q, k, w, out);
}
